// GAT_48017734369678
// MI455X (gfx1250) — matrix operand order repaired and checked by reading, not compiled, not run on hardware
//
#include <hip/hip_runtime.h>

// ============================================================================
// Dual-GAT + group attention for MI455X (gfx1250, wave32, WMMA 16x16x32 f16)
//
// v2: all WMMA fragments are built from contiguous 128-bit loads (B operands
// kept transposed so per-lane K-runs are contiguous); group-attention pass 2
// stages X2^T chunks into LDS with global_load_async_to_lds_b128 (ASYNCcnt),
// shared by all 8 waves of the workgroup.
// ============================================================================

typedef _Float16 h16;
typedef __attribute__((ext_vector_type(4)))  _Float16 v4h;
typedef __attribute__((ext_vector_type(8)))  _Float16 v8h;
typedef __attribute__((ext_vector_type(16))) _Float16 v16h;
typedef __attribute__((ext_vector_type(8)))  float    v8f;

#define NNODES 50000
#define NGRP   1024
#define NEDGE1 800000
#define NEDGE2 16384
#define FIN    128
#define NHEAD  4
#define NHID   32
#define NCLS   32

static __device__ __forceinline__ unsigned ordkey(float x) {
  unsigned u = __float_as_uint(x);
  return (u & 0x80000000u) ? ~u : (u | 0x80000000u);   // monotone float->uint
}
static __device__ __forceinline__ float orddec(unsigned k) {
  return (k & 0x80000000u) ? __uint_as_float(k ^ 0x80000000u) : __uint_as_float(~k);
}
static __device__ __forceinline__ v16h cat8(v8h lo, v8h hi) {
  return __builtin_shufflevector(lo, hi, 0, 1, 2, 3, 4, 5, 6, 7,
                                 8, 9, 10, 11, 12, 13, 14, 15);
}

// ---------------------------------------------------------------- utilities
__global__ void k_cvt_f16(const float* __restrict__ in, h16* __restrict__ out, long n4) {
  long i = (long)blockIdx.x * blockDim.x + threadIdx.x;   // one float4 per thread
  if (i >= n4) return;
  float4 v = ((const float4*)in)[i];
  v4h o = {(h16)v.x, (h16)v.y, (h16)v.z, (h16)v.w};
  ((v4h*)out)[i] = o;
}

// out[c*R + r] = (h16) in[r*Cc + c]   (build B^T operands)
__global__ void k_cvt_t(const float* __restrict__ in, h16* __restrict__ out,
                        int R, int Cc) {
  long i = (long)blockIdx.x * blockDim.x + threadIdx.x;
  if (i >= (long)R * Cc) return;
  int r = (int)(i / Cc), c = (int)(i - (long)r * Cc);
  out[(long)c * R + r] = (h16)in[i];
}

__global__ void k_zero(unsigned* __restrict__ p, long n) {
  long i = (long)blockIdx.x * blockDim.x + threadIdx.x;
  if (i < n) p[i] = 0u;
}

__global__ void k_build_A(const int* __restrict__ ei, float* __restrict__ A, int e2, int g) {
  int t = blockIdx.x * blockDim.x + threadIdx.x;
  if (t >= e2) return;
  int u = ei[t], v = ei[e2 + t];
  atomicAdd(&A[(long)u * g + v], 1.0f);
  if (u != v) atomicAdd(&A[(long)v * g + u], 1.0f);
}

// ------------------------------------------------------------- WMMA GEMM
// C(MxN) = A(MxK) * B(KxN); A f16 row-major, BT = B^T f16 (N x K row-major).
// One wave per 16x16 tile; fragments are contiguous b128 loads.
__global__ void __launch_bounds__(256) k_gemm_wmma(
    const h16* __restrict__ A, const h16* __restrict__ BT, float* __restrict__ C,
    int M, int Nc, int K) {
  int wave = threadIdx.x >> 5, lane = threadIdx.x & 31;
  int tn_cnt = Nc >> 4;
  int tile = blockIdx.x * (blockDim.x >> 5) + wave;
  if (tile >= (M >> 4) * tn_cnt) return;
  int tm = tile / tn_cnt, tn = tile - tm * tn_cnt;
  int am  = tm * 16 + (lane & 15);     // A-fragment: lane holds row m
  int kbA = (lane >> 4) * 16;          // K phase 0 / 16 per lane half
  int bn  = tn * 16 + (lane & 15);     // B-fragment: lane holds col n
  int kbB = (lane >> 4) * 16;          // K phase 0 / 16 per lane half
  const h16* arow = A + (long)am * K;
  const h16* brow = BT + (long)bn * K;
  v8f acc = {0.f, 0.f, 0.f, 0.f, 0.f, 0.f, 0.f, 0.f};
  for (int k0 = 0; k0 < K; k0 += 32) {
    v8h alo = *(const v8h*)(arow + k0 + kbA);
    v8h ahi = *(const v8h*)(arow + k0 + 8 + kbA);
    v16h bf = *(const v16h*)(brow + k0 + kbB);
    acc = __builtin_amdgcn_wmma_f32_16x16x32_f16(false, cat8(alo, ahi), false, bf,
                                                 (short)0, acc, false, false);
  }
  int row0 = tm * 16 + 8 * (lane >> 4);
#pragma unroll
  for (int j = 0; j < 8; ++j) C[(long)(row0 + j) * Nc + bn] = acc[j];
}

// --------------------------------------------------- GAT edge-softmax pieces
__global__ void k_alar(const float* __restrict__ feat,
                       const float* __restrict__ asrc, const float* __restrict__ adst,
                       float* __restrict__ al, float* __restrict__ ar,
                       int n, int H, int C) {
  int t = blockIdx.x * blockDim.x + threadIdx.x;
  if (t >= n * H) return;
  int node = t / H, h = t - node * H;
  int F = H * C;
  const float* fp = feat + (long)node * F + h * C;
  float sl = 0.f, sr = 0.f;
  for (int c = 0; c < C; ++c) {
    float v = fp[c];
    sl += v * asrc[h * C + c];
    sr += v * adst[h * C + c];
  }
  al[t] = sl; ar[t] = sr;
}

__global__ void k_edge_max(const int* __restrict__ ei, int E, int n, int H,
                           const float* __restrict__ al, const float* __restrict__ ar,
                           unsigned* __restrict__ mkey) {
  int t = blockIdx.x * blockDim.x + threadIdx.x;
  if (t >= E + n) return;
  int s, d;
  if (t < E) { s = ei[t]; d = ei[E + t]; } else { s = t - E; d = s; }
  for (int h = 0; h < H; ++h) {
    float v = al[s * H + h] + ar[d * H + h];
    v = v > 0.f ? v : 0.2f * v;
    atomicMax(&mkey[d * H + h], ordkey(v));
  }
}

__global__ void k_edge_sum(const int* __restrict__ ei, int E, int n, int H,
                           const float* __restrict__ al, const float* __restrict__ ar,
                           const unsigned* __restrict__ mkey, float* __restrict__ ssum) {
  int t = blockIdx.x * blockDim.x + threadIdx.x;
  if (t >= E + n) return;
  int s, d;
  if (t < E) { s = ei[t]; d = ei[E + t]; } else { s = t - E; d = s; }
  for (int h = 0; h < H; ++h) {
    float v = al[s * H + h] + ar[d * H + h];
    v = v > 0.f ? v : 0.2f * v;
    atomicAdd(&ssum[d * H + h], __expf(v - orddec(mkey[d * H + h])));
  }
}

__global__ void k_init_bias(float* __restrict__ out, const float* __restrict__ b,
                            int n, int F) {
  long i = (long)blockIdx.x * blockDim.x + threadIdx.x;
  if (i >= (long)n * F) return;
  out[i] = b[i % F];
}

// one wave per edge; lanes stride the feature dimension
__global__ void k_edge_scatter(const int* __restrict__ ei, int E, int n, int H, int C,
                               const float* __restrict__ al, const float* __restrict__ ar,
                               const unsigned* __restrict__ mkey, const float* __restrict__ ssum,
                               const float* __restrict__ feat, float* __restrict__ out) {
  int wave = threadIdx.x >> 5, lane = threadIdx.x & 31;
  int t = blockIdx.x * (blockDim.x >> 5) + wave;
  if (t >= E + n) return;
  int s, d;
  if (t < E) { s = ei[t]; d = ei[E + t]; } else { s = t - E; d = s; }
  int F = H * C;
  __builtin_prefetch(feat + (long)s * F, 0, 0);     // global_prefetch_b8
  for (int f = lane; f < F; f += 32) {
    int h = f / C;
    float v = al[s * H + h] + ar[d * H + h];
    v = v > 0.f ? v : 0.2f * v;
    float att = __expf(v - orddec(mkey[d * H + h])) / ssum[d * H + h];
    atomicAdd(&out[(long)d * F + f], att * feat[(long)s * F + f]);
  }
}

// ------------------------------------------------- fused group attention
// out[n] = sum_g w[n,g] X2[g] / sum_g w[n,g]
// w[n,g] = exp(S[n,g]-rowmax) * (A[gidx[n],g] + [g==gidx[n]]),  S = X*X2^T.
// One wave per 16-node block (tail waves clamp to the last block instead of
// returning, so workgroup barriers stay uniform). Pass 1: WMMA rowmax.
// Pass 2: per 32-group chunk, X2^T columns are staged into LDS with
// global_load_async_to_lds_b128 (shared by the 8 waves), S is recomputed,
// w is repacked C-layout -> A-fragment through a per-wave LDS slice, and the
// numerator accumulates via WMMA from the LDS-staged chunk.
template <int F>
__global__ void __launch_bounds__(256) k_group_attn(
    const h16* __restrict__ X,    // Nn x F  (row major)
    const h16* __restrict__ X2,   // Gg x F  (row major)
    const h16* __restrict__ X2T,  // F x Gg  (row major; = X2 transposed)
    const float* __restrict__ Amat, const int* __restrict__ gidx,
    float* __restrict__ grp, int Nn, int Gg) {
  constexpr int KF = F / 32;   // K fragments for S
  constexpr int OT = F / 16;   // output column tiles
  __shared__ __attribute__((aligned(16))) h16 ldsw[8][16 * 32];
  __shared__ __attribute__((aligned(16))) h16 ldsB[F * 32];
  int wave = threadIdx.x >> 5, lane = threadIdx.x & 31;
  int NB = Nn >> 4;
  int nb = blockIdx.x * (blockDim.x >> 5) + wave;
  if (nb >= NB) nb = NB - 1;           // clamp (redundant recompute, no deadlock)
  int n0 = nb * 16;
  int m   = lane & 15;
  int kbA = (lane >> 4) * 16;
  int nn  = lane & 15;
  int kbB = (lane >> 4) * 16;
  int rbase = 8 * (lane >> 4);

  // A-fragments of the 16 x F node block (resident in VGPRs)
  v16h xa[KF];
  const h16* xrow = X + (long)(n0 + m) * F;
#pragma unroll
  for (int kf = 0; kf < KF; ++kf) {
    v8h lo = *(const v8h*)(xrow + kf * 32 + kbA);
    v8h hi = *(const v8h*)(xrow + kf * 32 + 8 + kbA);
    xa[kf] = cat8(lo, hi);
  }
  int gID[8];
#pragma unroll
  for (int j = 0; j < 8; ++j) gID[j] = gidx[n0 + rbase + j];

  // ---- pass 1: row max of S over all G groups
  float rm[8];
#pragma unroll
  for (int j = 0; j < 8; ++j) rm[j] = -3.0e38f;
  for (int g0 = 0; g0 < Gg; g0 += 16) {
    v8f sc = {0.f, 0.f, 0.f, 0.f, 0.f, 0.f, 0.f, 0.f};
    const h16* b2row = X2 + (long)(g0 + nn) * F;
#pragma unroll
    for (int kf = 0; kf < KF; ++kf) {
      v16h bf = *(const v16h*)(b2row + kf * 32 + kbB);  // contiguous K run
      sc = __builtin_amdgcn_wmma_f32_16x16x32_f16(false, xa[kf], false, bf,
                                                  (short)0, sc, false, false);
    }
#pragma unroll
    for (int j = 0; j < 8; ++j) rm[j] = fmaxf(rm[j], sc[j]);
  }
#pragma unroll
  for (int off = 1; off < 16; off <<= 1)
#pragma unroll
    for (int j = 0; j < 8; ++j) rm[j] = fmaxf(rm[j], __shfl_xor(rm[j], off, 32));

  // ---- pass 2: weights + numerator/denominator
  v8f acc[OT];
#pragma unroll
  for (int t = 0; t < OT; ++t) acc[t] = (v8f){0.f, 0.f, 0.f, 0.f, 0.f, 0.f, 0.f, 0.f};
  float rd[8];
#pragma unroll
  for (int j = 0; j < 8; ++j) rd[j] = 0.f;

  for (int c0 = 0; c0 < Gg; c0 += 32) {
    __syncthreads();                   // previous chunk fully consumed
    // stage X2T[:, c0:c0+32] into ldsB[F][32] via async-to-LDS b128
    for (int i = threadIdx.x; i < F * 4; i += 256) {
      int r = i >> 2, c8 = (i & 3) * 8;
      unsigned dst = (unsigned)(unsigned long long)(const void*)&ldsB[r * 32 + c8];
      unsigned long long src =
          (unsigned long long)(const void*)(X2T + (long)r * Gg + c0 + c8);
      asm volatile("global_load_async_to_lds_b128 %0, %1, off"
                   :: "v"(dst), "v"(src) : "memory");
    }
    asm volatile("s_wait_asynccnt 0" ::: "memory");
    __syncthreads();

#pragma unroll
    for (int sub = 0; sub < 2; ++sub) {
      int g0 = c0 + sub * 16;
      v8f sc = {0.f, 0.f, 0.f, 0.f, 0.f, 0.f, 0.f, 0.f};
      const h16* b2row = X2 + (long)(g0 + nn) * F;
#pragma unroll
      for (int kf = 0; kf < KF; ++kf) {
        v16h bf = *(const v16h*)(b2row + kf * 32 + kbB);
        sc = __builtin_amdgcn_wmma_f32_16x16x32_f16(false, xa[kf], false, bf,
                                                    (short)0, sc, false, false);
      }
#pragma unroll
      for (int j = 0; j < 8; ++j) {
        int g = g0 + nn;
        float coef = Amat[(long)gID[j] * Gg + g] + ((g == gID[j]) ? 1.f : 0.f);
        float w = __expf(sc[j] - rm[j]) * coef;
        rd[j] += w;
        ldsw[wave][(rbase + j) * 32 + sub * 16 + nn] = (h16)w;   // plain (row,col)
      }
    }
    // repack w (16x32) from LDS into A-fragment layout (contiguous b128 reads)
    v8h wlo = *(const v8h*)&ldsw[wave][m * 32 + kbA];
    v8h whi = *(const v8h*)&ldsw[wave][m * 32 + 8 + kbA];
    v16h wf = cat8(wlo, whi);
    // numer += w(16x32) @ X2[c0:c0+32, :]   (B fragments from LDS-staged X2T)
#pragma unroll
    for (int t = 0; t < OT; ++t) {
      v16h bf = *(const v16h*)&ldsB[(t * 16 + nn) * 32 + kbB];
      acc[t] = __builtin_amdgcn_wmma_f32_16x16x32_f16(false, wf, false, bf,
                                                      (short)0, acc[t], false, false);
    }
  }
#pragma unroll
  for (int off = 1; off < 16; off <<= 1)
#pragma unroll
    for (int j = 0; j < 8; ++j) rd[j] += __shfl_xor(rd[j], off, 32);

#pragma unroll
  for (int t = 0; t < OT; ++t)
#pragma unroll
    for (int j = 0; j < 8; ++j)
      grp[(long)(n0 + rbase + j) * F + t * 16 + nn] = acc[t][j] / rd[j];
}

// ------------------------------------------- mix + layernorm (+elu) per node
__global__ void k_mix_ln(const float* __restrict__ ga, const float* __restrict__ gr,
                         const float* __restrict__ g, const float* __restrict__ b,
                         int n, int F, int do_elu,
                         float* __restrict__ out32, h16* __restrict__ out16) {
  int wave = threadIdx.x >> 5, lane = threadIdx.x & 31;
  int node = blockIdx.x * (blockDim.x >> 5) + wave;
  if (node >= n) return;
  float v[4];
  int cnt = F >> 5;
  float s = 0.f;
  for (int i = 0; i < cnt; ++i) {
    int f = lane + 32 * i;
    v[i] = 0.5f * (ga[(long)node * F + f] + gr[(long)node * F + f]);
    s += v[i];
  }
  for (int off = 16; off; off >>= 1) s += __shfl_xor(s, off, 32);
  float mu = s / (float)F;
  float q = 0.f;
  for (int i = 0; i < cnt; ++i) { float d = v[i] - mu; q += d * d; }
  for (int off = 16; off; off >>= 1) q += __shfl_xor(q, off, 32);
  float inv = rsqrtf(q / (float)F + 1e-5f);
  for (int i = 0; i < cnt; ++i) {
    int f = lane + 32 * i;
    float y = (v[i] - mu) * inv * g[f] + b[f];
    if (do_elu) y = y > 0.f ? y : (__expf(y) - 1.f);
    if (out32) out32[(long)node * F + f] = y;
    if (out16) out16[(long)node * F + f] = (h16)y;
  }
}

// ============================================================================
extern "C" void kernel_launch(void* const* d_in, const int* in_sizes, int n_in,
                              void* d_out, int out_size, void* d_ws, size_t ws_size,
                              hipStream_t stream) {
  (void)in_sizes; (void)n_in; (void)out_size; (void)ws_size;
  const int N = NNODES, G = NGRP, E = NEDGE1, E2 = NEDGE2;
  const int F = FIN, H = NHEAD, C = NHID, NC = NCLS;

  const float* x1   = (const float*)d_in[0];
  const int*   ei1  = (const int*)d_in[1];
  const float* x2   = (const float*)d_in[2];
  const int*   ei2  = (const int*)d_in[3];
  const int*   gidx = (const int*)d_in[4];
  const float* W1a  = (const float*)d_in[5];
  const float* a1as = (const float*)d_in[6];
  const float* a1ad = (const float*)d_in[7];
  const float* b1a  = (const float*)d_in[8];
  const float* W1b  = (const float*)d_in[9];
  const float* a1bs = (const float*)d_in[10];
  const float* a1bd = (const float*)d_in[11];
  const float* b1b  = (const float*)d_in[12];
  const float* ln1g = (const float*)d_in[13];
  const float* ln1b = (const float*)d_in[14];
  const float* W2a  = (const float*)d_in[15];
  const float* a2as = (const float*)d_in[16];
  const float* a2ad = (const float*)d_in[17];
  const float* b2a  = (const float*)d_in[18];
  const float* W2b  = (const float*)d_in[19];
  const float* a2bs = (const float*)d_in[20];
  const float* a2bd = (const float*)d_in[21];
  const float* b2b  = (const float*)d_in[22];
  const float* ln2g = (const float*)d_in[23];
  const float* ln2b = (const float*)d_in[24];

  char* wp = (char*)d_ws;
  auto alloc = [&](size_t bytes) -> void* {
    void* p = (void*)wp;
    wp += (bytes + 255) & ~(size_t)255;
    return p;
  };
  h16*  x1h    = (h16*)alloc((size_t)N * F * 2);
  h16*  x2h    = (h16*)alloc((size_t)G * F * 2);
  h16*  W1aT   = (h16*)alloc((size_t)F * F * 2);
  h16*  W1bT   = (h16*)alloc((size_t)F * F * 2);
  h16*  W2aT   = (h16*)alloc((size_t)F * NC * 2);
  h16*  W2bT   = (h16*)alloc((size_t)F * NC * 2);
  float* Amat  = (float*)alloc((size_t)G * G * 4);
  float* feat1 = (float*)alloc((size_t)N * F * 4);
  float* featg = (float*)alloc((size_t)G * F * 4);
  float* al1   = (float*)alloc((size_t)N * H * 4);
  float* ar1   = (float*)alloc((size_t)N * H * 4);
  float* alg   = (float*)alloc((size_t)G * H * 4);
  float* arg   = (float*)alloc((size_t)G * H * 4);
  unsigned* mk1 = (unsigned*)alloc((size_t)N * H * 4);
  float*    ss1 = (float*)alloc((size_t)N * H * 4);
  unsigned* mkg = (unsigned*)alloc((size_t)G * H * 4);
  float*    ssg = (float*)alloc((size_t)G * H * 4);
  float* gat1  = (float*)alloc((size_t)N * F * 4);
  float* gatg  = (float*)alloc((size_t)G * F * 4);
  h16*  gat1h  = (h16*)alloc((size_t)N * F * 2);
  h16*  gatgh  = (h16*)alloc((size_t)G * F * 2);
  h16*  gatgT  = (h16*)alloc((size_t)G * F * 2);   // F x G
  float* grp1  = (float*)alloc((size_t)N * F * 4);
  h16*  h1h    = (h16*)alloc((size_t)N * F * 2);
  float* feat2 = (float*)alloc((size_t)N * NC * 4);
  float* featg2= (float*)alloc((size_t)G * NC * 4);
  float* al2   = (float*)alloc((size_t)N * 4);
  float* ar2   = (float*)alloc((size_t)N * 4);
  float* alg2  = (float*)alloc((size_t)G * 4);
  float* arg2  = (float*)alloc((size_t)G * 4);
  unsigned* mk2  = (unsigned*)alloc((size_t)N * 4);
  float*    ss2  = (float*)alloc((size_t)N * 4);
  unsigned* mkg2 = (unsigned*)alloc((size_t)G * 4);
  float*    ssg2 = (float*)alloc((size_t)G * 4);
  float* gat2  = (float*)alloc((size_t)N * NC * 4);
  float* gatg2 = (float*)alloc((size_t)G * NC * 4);
  h16*  gat2h  = (h16*)alloc((size_t)N * NC * 2);
  h16*  gatg2h = (h16*)alloc((size_t)G * NC * 2);
  h16*  gatg2T = (h16*)alloc((size_t)G * NC * 2);  // NC x G
  float* grp2  = (float*)alloc((size_t)N * NC * 4);

  const int TB = 256;
  auto blk1 = [](long n) { return (unsigned)((n + 255) / 256); };
  auto blkw = [](long waves) { return (unsigned)((waves + 7) / 8); };

  // conversions (vectorized, element counts are multiples of 4)
  k_cvt_f16<<<blk1((long)N * F / 4), TB, 0, stream>>>(x1, x1h, (long)N * F / 4);
  k_cvt_f16<<<blk1((long)G * F / 4), TB, 0, stream>>>(x2, x2h, (long)G * F / 4);
  // transposed-converted B operands for the GEMMs
  k_cvt_t<<<blk1((long)F * F), TB, 0, stream>>>(W1a, W1aT, F, F);
  k_cvt_t<<<blk1((long)F * F), TB, 0, stream>>>(W1b, W1bT, F, F);
  k_cvt_t<<<blk1((long)F * NC), TB, 0, stream>>>(W2a, W2aT, F, NC);
  k_cvt_t<<<blk1((long)F * NC), TB, 0, stream>>>(W2b, W2bT, F, NC);

  // adjacency multiplicity matrix
  k_zero<<<blk1((long)G * G), TB, 0, stream>>>((unsigned*)Amat, (long)G * G);
  k_build_A<<<blk1(E2), TB, 0, stream>>>(ei2, Amat, E2, G);

  // ---- layer 1 dense projections (WMMA)
  k_gemm_wmma<<<blkw((long)(N / 16) * (F / 16)), TB, 0, stream>>>(x1h, W1aT, feat1, N, F, F);
  k_gemm_wmma<<<blkw((long)(G / 16) * (F / 16)), TB, 0, stream>>>(x2h, W1bT, featg, G, F, F);

  // ---- layer 1 GAT edge softmax (graph 1)
  k_alar<<<blk1((long)N * H), TB, 0, stream>>>(feat1, a1as, a1ad, al1, ar1, N, H, C);
  k_zero<<<blk1((long)N * H), TB, 0, stream>>>(mk1, (long)N * H);
  k_zero<<<blk1((long)N * H), TB, 0, stream>>>((unsigned*)ss1, (long)N * H);
  k_edge_max<<<blk1((long)E + N), TB, 0, stream>>>(ei1, E, N, H, al1, ar1, mk1);
  k_edge_sum<<<blk1((long)E + N), TB, 0, stream>>>(ei1, E, N, H, al1, ar1, mk1, ss1);
  k_init_bias<<<blk1((long)N * F), TB, 0, stream>>>(gat1, b1a, N, F);
  k_edge_scatter<<<blkw((long)E + N), TB, 0, stream>>>(ei1, E, N, H, C, al1, ar1, mk1, ss1, feat1, gat1);

  // ---- layer 1 GAT edge softmax (group graph)
  k_alar<<<blk1((long)G * H), TB, 0, stream>>>(featg, a1bs, a1bd, alg, arg, G, H, C);
  k_zero<<<blk1((long)G * H), TB, 0, stream>>>(mkg, (long)G * H);
  k_zero<<<blk1((long)G * H), TB, 0, stream>>>((unsigned*)ssg, (long)G * H);
  k_edge_max<<<blk1((long)E2 + G), TB, 0, stream>>>(ei2, E2, G, H, alg, arg, mkg);
  k_edge_sum<<<blk1((long)E2 + G), TB, 0, stream>>>(ei2, E2, G, H, alg, arg, mkg, ssg);
  k_init_bias<<<blk1((long)G * F), TB, 0, stream>>>(gatg, b1b, G, F);
  k_edge_scatter<<<blkw((long)E2 + G), TB, 0, stream>>>(ei2, E2, G, H, C, alg, arg, mkg, ssg, featg, gatg);

  // ---- layer 1 fused group attention (WMMA flash-style, async-LDS staging)
  k_cvt_f16<<<blk1((long)N * F / 4), TB, 0, stream>>>(gat1, gat1h, (long)N * F / 4);
  k_cvt_f16<<<blk1((long)G * F / 4), TB, 0, stream>>>(gatg, gatgh, (long)G * F / 4);
  k_cvt_t<<<blk1((long)G * F), TB, 0, stream>>>(gatg, gatgT, G, F);
  k_group_attn<FIN><<<blkw((long)N / 16), TB, 0, stream>>>(gat1h, gatgh, gatgT, Amat, gidx, grp1, N, G);

  // ---- mix + LN + ELU -> h1 (f16 for next layer)
  k_mix_ln<<<blkw(N), TB, 0, stream>>>(gat1, grp1, ln1g, ln1b, N, F, 1, nullptr, h1h);

  // ---- layer 2 dense projections (WMMA). Group-graph input is raw gatg.
  k_gemm_wmma<<<blkw((long)(N / 16) * (NC / 16)), TB, 0, stream>>>(h1h, W2aT, feat2, N, NC, F);
  k_gemm_wmma<<<blkw((long)(G / 16) * (NC / 16)), TB, 0, stream>>>(gatgh, W2bT, featg2, G, NC, F);

  // ---- layer 2 GAT (graph 1, H=1, C=32)
  k_alar<<<blk1((long)N), TB, 0, stream>>>(feat2, a2as, a2ad, al2, ar2, N, 1, NC);
  k_zero<<<blk1((long)N), TB, 0, stream>>>(mk2, (long)N);
  k_zero<<<blk1((long)N), TB, 0, stream>>>((unsigned*)ss2, (long)N);
  k_edge_max<<<blk1((long)E + N), TB, 0, stream>>>(ei1, E, N, 1, al2, ar2, mk2);
  k_edge_sum<<<blk1((long)E + N), TB, 0, stream>>>(ei1, E, N, 1, al2, ar2, mk2, ss2);
  k_init_bias<<<blk1((long)N * NC), TB, 0, stream>>>(gat2, b2a, N, NC);
  k_edge_scatter<<<blkw((long)E + N), TB, 0, stream>>>(ei1, E, N, 1, NC, al2, ar2, mk2, ss2, feat2, gat2);

  // ---- layer 2 GAT (group graph)
  k_alar<<<blk1((long)G), TB, 0, stream>>>(featg2, a2bs, a2bd, alg2, arg2, G, 1, NC);
  k_zero<<<blk1((long)G), TB, 0, stream>>>(mkg2, (long)G);
  k_zero<<<blk1((long)G), TB, 0, stream>>>((unsigned*)ssg2, (long)G);
  k_edge_max<<<blk1((long)E2 + G), TB, 0, stream>>>(ei2, E2, G, 1, alg2, arg2, mkg2);
  k_edge_sum<<<blk1((long)E2 + G), TB, 0, stream>>>(ei2, E2, G, 1, alg2, arg2, mkg2, ssg2);
  k_init_bias<<<blk1((long)G * NC), TB, 0, stream>>>(gatg2, b2b, G, NC);
  k_edge_scatter<<<blkw((long)E2 + G), TB, 0, stream>>>(ei2, E2, G, 1, NC, alg2, arg2, mkg2, ssg2, featg2, gatg2);

  // ---- layer 2 fused group attention
  k_cvt_f16<<<blk1((long)N * NC / 4), TB, 0, stream>>>(gat2, gat2h, (long)N * NC / 4);
  k_cvt_f16<<<blk1((long)G * NC / 4), TB, 0, stream>>>(gatg2, gatg2h, (long)G * NC / 4);
  k_cvt_t<<<blk1((long)G * NC), TB, 0, stream>>>(gatg2, gatg2T, G, NC);
  k_group_attn<NCLS><<<blkw((long)N / 16), TB, 0, stream>>>(gat2h, gatg2h, gatg2T, Amat, gidx, grp2, N, G);

  // ---- final mix + LN -> d_out (f32, N x 32)
  k_mix_ln<<<blkw(N), TB, 0, stream>>>(gat2, grp2, ln2g, ln2b, N, NC, 0, (float*)d_out, nullptr);
}
